// muilt_view_encode_10242_5626407157948
// MI455X (gfx1250) — matrix operand order repaired and checked by reading, not compiled, not run on hardware
//
#include <hip/hip_runtime.h>

typedef __attribute__((ext_vector_type(16))) _Float16 v16h;
typedef __attribute__((ext_vector_type(8)))  _Float16 v8h;
typedef __attribute__((ext_vector_type(8)))  float    v8f;

#define NEG_SLOPE 0.2f

__device__ __forceinline__ int imin(int a, int b) { return a < b ? a : b; }

// ---------------------------------------------------------------------------
// Elementwise / prep kernels
// ---------------------------------------------------------------------------

// x: (B, C, N) f32 -> out: (B, N, C) f16
__global__ void k_transpose_in(const float* __restrict__ x, _Float16* __restrict__ out,
                               int Bn, int C, int N) {
  long i = (long)blockIdx.x * blockDim.x + threadIdx.x;
  long total = (long)Bn * N * C;
  if (i >= total) return;
  int c = (int)(i % C);
  long r = i / C;
  int p = (int)(r % N);
  int b = (int)(r / N);
  out[i] = (_Float16)x[((long)b * C + c) * N + p];
}

// W: (K, Nn) f32 row-major -> WT: (Nn, Kpad) f16 row-major, zero padded K..Kpad
__global__ void k_prep_wT(const float* __restrict__ W, _Float16* __restrict__ WT,
                          int K, int Nn, int Kpad) {
  int i = blockIdx.x * blockDim.x + threadIdx.x;
  if (i >= Nn * Kpad) return;
  int n = i / Kpad, k = i % Kpad;
  WT[i] = (k < K) ? (_Float16)W[(long)k * Nn + n] : (_Float16)0.0f;
}

// ---------------------------------------------------------------------------
// Gather-conv WMMA GEMM:  Y[b,p,n] = sum_k A[b,p,k] * W[k,n] + bias[n]
// where A[b,p, t*Cin+ci] = X[b, no[7p+t], ci]  (K zero-padded to Kpad)
// One wave computes a 16(M) x 32(N) tile: two 16x16 WMMA accumulators share a
// single gathered A fragment (halves gather traffic, doubles WMMA density).
// 4 waves/block tile M by 64.  Requires Cout % 32 == 0 (true for all layers).
// ---------------------------------------------------------------------------
__global__ __launch_bounds__(128)
void k_gconv_wmma(const _Float16* __restrict__ X, const int* __restrict__ no,
                  const _Float16* __restrict__ WT, const float* __restrict__ bias,
                  float* __restrict__ Y,
                  int Npts, int Cin, int cinLog2, int Cout, int Kpad) {
  const int lane  = threadIdx.x & 31;
  const int wave  = threadIdx.x >> 5;
  const int mbase = blockIdx.x * 64 + wave * 16;
  const int nbase = blockIdx.y * 32;
  const int b     = blockIdx.z;
  if (mbase >= Npts) return;

  const int row = lane & 15;
  const int hi  = lane >> 4;     // 0 for lanes 0-15, 1 for lanes 16-31
  const int p   = imin(mbase + row, Npts - 1);

  int g[7];
#pragma unroll
  for (int t = 0; t < 7; ++t) g[t] = no[7 * p + t];

  const _Float16* Xb    = X + (size_t)b * Npts * Cin;
  const int       n0    = nbase + row;            // output columns (Cout % 32 == 0)
  const int       n1    = n0 + 16;
  const _Float16* Bcol0 = WT + (size_t)n0 * Kpad; // contiguous K for these columns
  const _Float16* Bcol1 = WT + (size_t)n1 * Kpad;
  const int kgA = hi * 8;    // A: low lanes K {0..7,16..23}, high lanes {8..15,24..31}
  const int kbB = hi * 8;    // B: low lanes K {0..7,16..23}, high lanes {8..15,24..31}
  const int KA  = 7 * Cin;

  v8f acc0 = {0.f, 0.f, 0.f, 0.f, 0.f, 0.f, 0.f, 0.f};
  v8f acc1 = {0.f, 0.f, 0.f, 0.f, 0.f, 0.f, 0.f, 0.f};

  for (int k0 = 0; k0 < Kpad; k0 += 32) {
    v16h a;
    if (cinLog2 >= 0) {
      // Cin is a power-of-two multiple of 8: each 8-half span stays in one neighbor
      int kb0 = k0 + kgA;
      int kb1 = kb0 + 16;
      int nb0 = kb0 >> cinLog2, ci0 = kb0 & (Cin - 1);
      int nb1 = kb1 >> cinLog2, ci1 = kb1 & (Cin - 1);
      v8h a0 = *(const v8h*)(Xb + (size_t)g[nb0] * Cin + ci0);
      v8h a1 = *(const v8h*)(Xb + (size_t)g[nb1] * Cin + ci1);
#pragma unroll
      for (int h = 0; h < 8; ++h) { a[h] = a0[h]; a[h + 8] = a1[h]; }
    } else {
      // First layer only: Cin == 3, KA == 21, Kpad == 32
#pragma unroll
      for (int h = 0; h < 16; ++h) {
        int kk = k0 + kgA + h + (h < 8 ? 0 : 8);
        _Float16 v = (_Float16)0.0f;
        if (kk < KA) {
          unsigned u  = (unsigned)kk;
          unsigned nb = u / 3u;
          unsigned ci = u - nb * 3u;
          v = Xb[(size_t)g[nb] * 3u + ci];
        }
        a[h] = v;
      }
    }
    v8h b00 = *(const v8h*)(Bcol0 + k0 + kbB);
    v8h b01 = *(const v8h*)(Bcol0 + k0 + kbB + 16);
    v8h b10 = *(const v8h*)(Bcol1 + k0 + kbB);
    v8h b11 = *(const v8h*)(Bcol1 + k0 + kbB + 16);
    v16h bb0, bb1;
#pragma unroll
    for (int h = 0; h < 8; ++h) {
      bb0[h] = b00[h]; bb0[h + 8] = b01[h];
      bb1[h] = b10[h]; bb1[h + 8] = b11[h];
    }
    acc0 = __builtin_amdgcn_wmma_f32_16x16x32_f16(false, a, false, bb0,
                                                  (short)0, acc0, false, false);
    acc1 = __builtin_amdgcn_wmma_f32_16x16x32_f16(false, a, false, bb1,
                                                  (short)0, acc1, false, false);
  }

  const float bn0 = bias ? bias[n0] : 0.f;
  const float bn1 = bias ? bias[n1] : 0.f;
#pragma unroll
  for (int j = 0; j < 8; ++j) {
    int pm = mbase + j + hi * 8;   // D layout: VGPR j -> row j (+8 for high lanes)
    if (pm < Npts) {
      float* Yr = Y + ((size_t)b * Npts + pm) * Cout;
      Yr[n0] = acc0[j] + bn0;
      Yr[n1] = acc1[j] + bn1;
    }
  }
}

// ---------------------------------------------------------------------------
// BatchNorm statistics: per-channel scale/shift so y = x*scale[c] + shift[c]
// (deterministic tree reduction -- no float atomics, safe for graph replay)
// ---------------------------------------------------------------------------
__global__ void k_bn_stats(const float* __restrict__ Y, long rows, int C,
                           const float* __restrict__ g, const float* __restrict__ be,
                           float* __restrict__ stats) {
  const int c = blockIdx.x, tid = threadIdx.x;
  float s = 0.f, ss = 0.f;
  for (long r = tid; r < rows; r += blockDim.x) {
    float x = Y[r * C + c];
    s += x; ss += x * x;
  }
  __shared__ float s1[256], s2[256];
  s1[tid] = s; s2[tid] = ss;
  __syncthreads();
  for (int st = 128; st > 0; st >>= 1) {
    if (tid < st) { s1[tid] += s1[tid + st]; s2[tid] += s2[tid + st]; }
    __syncthreads();
  }
  if (tid == 0) {
    float m  = s1[0] / (float)rows;
    float v  = s2[0] / (float)rows - m * m;
    float sc = g[c] * rsqrtf(v + 1e-5f);
    stats[c]       = sc;
    stats[256 + c] = be[c] - m * sc;
  }
}

// BN apply + LeakyReLU, f32 -> f16
__global__ void k_bn_apply(const float* __restrict__ Y, const float* __restrict__ stats,
                           _Float16* __restrict__ out, long total, int C) {
  long i = (long)blockIdx.x * blockDim.x + threadIdx.x;
  if (i >= total) return;
  int c = (int)(i % C);
  float y = Y[i] * stats[c] + stats[256 + c];
  y = (y >= 0.f) ? y : NEG_SLOPE * y;
  out[i] = (_Float16)y;
}

// 7-neighbor average pooling, f16 -> f16
__global__ void k_pool7(const _Float16* __restrict__ in, const int* __restrict__ no,
                        _Float16* __restrict__ out, int nf, int nc, int C, long total) {
  long i = (long)blockIdx.x * blockDim.x + threadIdx.x;
  if (i >= total) return;
  int c = (int)(i % C);
  long r = i / C;
  int j = (int)(r % nc);
  int b = (int)(r / nc);
  const _Float16* inb = in + (long)b * nf * C;
  float s = 0.f;
#pragma unroll
  for (int t = 0; t < 7; ++t) { int gg = no[7 * j + t]; s += (float)inb[(long)gg * C + c]; }
  out[i] = (_Float16)(s * (1.0f / 7.0f));
}

// Copy encoder output (B, Ntok, 256) into xc at token offset, both f16 and f32
__global__ void k_copy_xc(const _Float16* __restrict__ act, _Float16* __restrict__ xc16,
                          float* __restrict__ xc32, int toff, int Ntok, int T, long total) {
  long i = (long)blockIdx.x * blockDim.x + threadIdx.x;
  if (i >= total) return;
  int c = (int)(i % 256);
  long r = i / 256;
  int t = (int)(r % Ntok);
  int b = (int)(r / Ntok);
  long dst = ((long)b * T + toff + t) * 256 + c;
  float v = (float)act[i];
  xc16[dst] = (_Float16)v;
  xc32[dst] = v;
}

// ---------------------------------------------------------------------------
// Generic batched WMMA GEMM:  C = A @ B + bias, with B supplied transposed
// A:  (M, lda) f16 row-major, batch stride sA
// BT: (Nn, ldb) f16 row-major (i.e. B^T), batch stride sB (0 => shared weights)
// C:  f32 or f16, normal (M, ldc) or transposed (Nn, ldc), batch stride sC
// Handles M/N/K edges (clamped loads, guarded K remainder, masked stores).
// ---------------------------------------------------------------------------
#define GEMM_F16OUT 1
#define GEMM_TRANS  2

__global__ __launch_bounds__(128)
void k_gemm_wmma(const _Float16* __restrict__ A, const _Float16* __restrict__ BT,
                 const float* __restrict__ bias, void* __restrict__ Cptr,
                 int M, int Nn, int K, int lda, int ldb, int ldc,
                 long sA, long sB, long sC, int flags) {
  const int lane  = threadIdx.x & 31;
  const int wave  = threadIdx.x >> 5;
  const int mbase = blockIdx.x * 64 + wave * 16;
  const int nbase = blockIdx.y * 16;
  const int b     = blockIdx.z;
  if (mbase >= M) return;

  const int row = lane & 15;
  const int hi  = lane >> 4;
  const int rA  = imin(mbase + row, M - 1);
  const int rB  = imin(nbase + row, Nn - 1);
  const _Float16* Arow = A  + (long)b * sA + (long)rA * lda;
  const _Float16* Brow = BT + (long)b * sB + (long)rB * ldb;
  const int kgA = hi * 8;
  const int kbB = hi * 8;

  v8f acc = {0.f, 0.f, 0.f, 0.f, 0.f, 0.f, 0.f, 0.f};

  int k0 = 0;
  for (; k0 + 32 <= K; k0 += 32) {
    v8h a0 = *(const v8h*)(Arow + k0 + kgA);
    v8h a1 = *(const v8h*)(Arow + k0 + kgA + 16);
    v8h b0 = *(const v8h*)(Brow + k0 + kbB);
    v8h b1 = *(const v8h*)(Brow + k0 + kbB + 16);
    v16h a, bb;
#pragma unroll
    for (int h = 0; h < 8; ++h) {
      a[h] = a0[h];  a[h + 8]  = a1[h];
      bb[h] = b0[h]; bb[h + 8] = b1[h];
    }
    acc = __builtin_amdgcn_wmma_f32_16x16x32_f16(false, a, false, bb,
                                                 (short)0, acc, false, false);
  }
  if (k0 < K) {  // K remainder, element-guarded (zeros beyond K)
    v16h a, bb;
#pragma unroll
    for (int h = 0; h < 16; ++h) {
      int ka = k0 + kgA + h + (h < 8 ? 0 : 8);
      int kb = k0 + kbB + h + (h < 8 ? 0 : 8);
      a[h]  = (ka < K) ? Arow[ka] : (_Float16)0.0f;
      bb[h] = (kb < K) ? Brow[kb] : (_Float16)0.0f;
    }
    acc = __builtin_amdgcn_wmma_f32_16x16x32_f16(false, a, false, bb,
                                                 (short)0, acc, false, false);
  }

  const int cn = nbase + row;
  const float bn = (bias && cn < Nn) ? bias[cn] : 0.f;
#pragma unroll
  for (int j = 0; j < 8; ++j) {
    int pm = mbase + j + hi * 8;
    if (pm < M && cn < Nn) {
      float v = acc[j] + bn;
      long idx = (flags & GEMM_TRANS) ? ((long)cn * ldc + pm) : ((long)pm * ldc + cn);
      if (flags & GEMM_F16OUT) ((_Float16*)Cptr)[(long)b * sC + idx] = (_Float16)v;
      else                     ((float*)Cptr)[(long)b * sC + idx]    = v;
    }
  }
}

// Row softmax (len 324), f32 in -> f16 out with padded row pitch
__global__ void k_softmax_rows(const float* __restrict__ S, _Float16* __restrict__ Aout,
                               int len, int ldo) {
  __shared__ float red[128];
  const int r = blockIdx.x, tid = threadIdx.x;
  const float* src = S + (long)r * len;
  float mx = -3.4e38f;
  for (int i = tid; i < len; i += 128) mx = fmaxf(mx, src[i]);
  red[tid] = mx; __syncthreads();
  for (int st = 64; st > 0; st >>= 1) {
    if (tid < st) red[tid] = fmaxf(red[tid], red[tid + st]);
    __syncthreads();
  }
  mx = red[0]; __syncthreads();
  float s = 0.f;
  for (int i = tid; i < len; i += 128) s += __expf(src[i] - mx);
  red[tid] = s; __syncthreads();
  for (int st = 64; st > 0; st >>= 1) {
    if (tid < st) red[tid] += red[tid + st];
    __syncthreads();
  }
  float inv = 1.f / red[0];
  _Float16* dst = Aout + (long)r * ldo;
  for (int i = tid; i < len; i += 128) dst[i] = (_Float16)(__expf(src[i] - mx) * inv);
}

// x2a = gamma * (attn@v) + xc
__global__ void k_x2a(const float* __restrict__ av, const float* __restrict__ xc32,
                      const float* __restrict__ gamma, float* __restrict__ out, long total) {
  long i = (long)blockIdx.x * blockDim.x + threadIdx.x;
  if (i >= total) return;
  out[i] = gamma[0] * av[i] + xc32[i];
}

// Global average pool over tokens + 256->2 linear + softmax
__global__ void k_classifier(const float* __restrict__ x2a, const float* __restrict__ wc,
                             const float* __restrict__ bc, float* __restrict__ out, int T) {
  const int b = blockIdx.x, tid = threadIdx.x;   // 256 threads = 256 channels
  const float* base = x2a + (long)b * T * 256;
  float s = 0.f;
  for (int t = 0; t < T; ++t) s += base[(long)t * 256 + tid];
  float gap = s / (float)T;
  __shared__ float red[256];
  __shared__ float logit[2];
  for (int j = 0; j < 2; ++j) {
    red[tid] = gap * wc[tid * 2 + j];
    __syncthreads();
    for (int st = 128; st > 0; st >>= 1) {
      if (tid < st) red[tid] += red[tid + st];
      __syncthreads();
    }
    if (tid == 0) logit[j] = red[0] + bc[j];
    __syncthreads();
  }
  if (tid == 0) {
    float m  = fmaxf(logit[0], logit[1]);
    float e0 = __expf(logit[0] - m), e1 = __expf(logit[1] - m);
    float inv = 1.f / (e0 + e1);
    out[b * 2 + 0] = e0 * inv;
    out[b * 2 + 1] = e1 * inv;
  }
}

// ---------------------------------------------------------------------------
// Host driver
// ---------------------------------------------------------------------------
extern "C" void kernel_launch(void* const* d_in, const int* in_sizes, int n_in,
                              void* d_out, int out_size, void* d_ws, size_t ws_size,
                              hipStream_t stream) {
  (void)in_sizes; (void)n_in; (void)out_size;
  const int B = 32, N3 = 10242, N4 = 2562, N5 = 642, N6 = 162, T = 324;

  const float* x1  = (const float*)d_in[0];
  const float* x2  = (const float*)d_in[1];
  const int*   no2 = (const int*)d_in[2];
  const int*   no3 = (const int*)d_in[3];
  const int*   no4 = (const int*)d_in[4];
  const int*   no5 = (const int*)d_in[5];
  auto f = [&](int i) { return (const float*)d_in[i]; };

  char* ws = (char*)d_ws;
  size_t cur = 0;
  auto alloc = [&](size_t bytes) -> char* {
    char* p = ws + cur;
    cur = (cur + bytes + 255) & ~(size_t)255;
    return p;
  };

  // ---- weight transposition (f32 (K,N) -> f16 (N,Kpad)) ----
  struct WInfo { int K, N, Kpad, inIdx; };
  const WInfo wi[11] = {
    {  21,  32,   32,  6}, { 224,  32,  224, 10},   // d3 w1, w2
    { 224,  64,  224, 14}, { 448,  64,  448, 18},   // d4
    { 448, 128,  448, 22}, { 896, 128,  896, 26},   // d5
    { 896, 256,  896, 30}, {1792, 256, 1792, 34},   // d6
    { 256,  32,  256, 38}, { 256,  32,  256, 40},   // wq, wk
    { 256, 256,  256, 42}                            // wv
  };
  _Float16* wT[11];
  for (int i = 0; i < 11; ++i) {
    wT[i] = (_Float16*)alloc((size_t)wi[i].N * wi[i].Kpad * 2);
    int tot = wi[i].N * wi[i].Kpad;
    k_prep_wT<<<(tot + 255) / 256, 256, 0, stream>>>(f(wi[i].inIdx), wT[i],
                                                     wi[i].K, wi[i].N, wi[i].Kpad);
  }

  // ---- scratch buffers ----
  _Float16* actA    = (_Float16*)alloc((size_t)B * N3 * 32 * 2);
  _Float16* actB    = (_Float16*)alloc((size_t)B * N3 * 32 * 2);
  float*    convbuf = (float*)alloc((size_t)B * N3 * 32 * 4);   // also scores / attnv
  float*    stats   = (float*)alloc(512 * 4);
  _Float16* xc16    = (_Float16*)alloc((size_t)B * T * 256 * 2);
  float*    xc32    = (float*)alloc((size_t)B * T * 256 * 4);
  _Float16* q16     = (_Float16*)alloc((size_t)B * T * 32 * 2);
  _Float16* k16     = (_Float16*)alloc((size_t)B * T * 32 * 2);
  _Float16* vT16    = (_Float16*)alloc((size_t)B * 256 * 336 * 2); // padded K pitch
  _Float16* attn16  = (_Float16*)alloc((size_t)B * T * 336 * 2);   // padded K pitch
  if (cur > ws_size) return;  // insufficient workspace (deterministic no-op)

  float* scoresF = convbuf;                           // 32*324*324 f32 (13.4 MB)
  float* attnvF  = convbuf + (size_t)4 * 1024 * 1024; // 32*324*256 f32 (10.6 MB)

  auto convbn = [&](const _Float16* inA, _Float16* outA, const int* no, int Npts,
                    int Cin, int cinLog2, int Cout, int Kpad, _Float16* WTp,
                    const float* bias, const float* gam, const float* bet) {
    dim3 grid((Npts + 63) / 64, Cout / 32, B);
    k_gconv_wmma<<<grid, 128, 0, stream>>>(inA, no, WTp, bias, convbuf,
                                           Npts, Cin, cinLog2, Cout, Kpad);
    k_bn_stats<<<Cout, 256, 0, stream>>>(convbuf, (long)B * Npts, Cout, gam, bet, stats);
    long tot = (long)B * Npts * Cout;
    k_bn_apply<<<(int)((tot + 255) / 256), 256, 0, stream>>>(convbuf, stats, outA, tot, Cout);
  };
  auto pool = [&](const _Float16* inA, _Float16* outA, const int* no, int nf, int nc, int C) {
    long tot = (long)B * nc * C;
    k_pool7<<<(int)((tot + 255) / 256), 256, 0, stream>>>(inA, no, outA, nf, nc, C, tot);
  };

  // ---- encoders (weight-shared; run sequentially, stats are per-encoder) ----
  for (int e = 0; e < 2; ++e) {
    const float* xin = e ? x2 : x1;
    long tin = (long)B * N3 * 3;
    k_transpose_in<<<(int)((tin + 255) / 256), 256, 0, stream>>>(xin, actA, B, 3, N3);

    convbn(actA, actB, no2, N3,   3, -1,  32,   32, wT[0], f(7),  f(8),  f(9));
    convbn(actB, actA, no2, N3,  32,  5,  32,  224, wT[1], f(11), f(12), f(13));
    pool  (actA, actB, no2, N3, N4, 32);
    convbn(actB, actA, no3, N4,  32,  5,  64,  224, wT[2], f(15), f(16), f(17));
    convbn(actA, actB, no3, N4,  64,  6,  64,  448, wT[3], f(19), f(20), f(21));
    pool  (actB, actA, no3, N4, N5, 64);
    convbn(actA, actB, no4, N5,  64,  6, 128,  448, wT[4], f(23), f(24), f(25));
    convbn(actB, actA, no4, N5, 128,  7, 128,  896, wT[5], f(27), f(28), f(29));
    pool  (actA, actB, no4, N5, N6, 128);
    convbn(actB, actA, no5, N6, 128,  7, 256,  896, wT[6], f(31), f(32), f(33));
    convbn(actA, actB, no5, N6, 256,  8, 256, 1792, wT[7], f(35), f(36), f(37));

    long tc = (long)B * N6 * 256;
    k_copy_xc<<<(int)((tc + 255) / 256), 256, 0, stream>>>(actB, xc16, xc32,
                                                           e * N6, N6, T, tc);
  }

  // ---- self-attention ----
  {  // q = xc @ wq + bq  -> f16 (B, 324, 32)
    dim3 g((T + 63) / 64, 2, B);
    k_gemm_wmma<<<g, 128, 0, stream>>>(xc16, wT[8], f(39), q16,
        T, 32, 256, 256, 256, 32, (long)T * 256, 0, (long)T * 32, GEMM_F16OUT);
  }
  {  // k = xc @ wk + bk  -> f16 (B, 324, 32)
    dim3 g((T + 63) / 64, 2, B);
    k_gemm_wmma<<<g, 128, 0, stream>>>(xc16, wT[9], f(41), k16,
        T, 32, 256, 256, 256, 32, (long)T * 256, 0, (long)T * 32, GEMM_F16OUT);
  }
  {  // v = xc @ wv + bv  -> f16 TRANSPOSED (B, 256, 336pitch) for reuse as B^T
    dim3 g((T + 63) / 64, 16, B);
    k_gemm_wmma<<<g, 128, 0, stream>>>(xc16, wT[10], f(43), vT16,
        T, 256, 256, 256, 256, 336, (long)T * 256, 0, (long)256 * 336,
        GEMM_F16OUT | GEMM_TRANS);
  }
  {  // scores = q @ k^T  -> f32 (B, 324, 324); B^T is exactly k's storage
    dim3 g((T + 63) / 64, (T + 15) / 16, B);
    k_gemm_wmma<<<g, 128, 0, stream>>>(q16, k16, nullptr, scoresF,
        T, T, 32, 32, 32, T, (long)T * 32, (long)T * 32, (long)T * T, 0);
  }
  k_softmax_rows<<<B * T, 128, 0, stream>>>(scoresF, attn16, T, 336);
  {  // attnv = attn @ v  -> f32 (B, 324, 256), K=324 with guarded remainder
    dim3 g((T + 63) / 64, 16, B);
    k_gemm_wmma<<<g, 128, 0, stream>>>(attn16, vT16, nullptr, attnvF,
        T, 256, T, 336, 336, 256, (long)T * 336, (long)256 * 336, (long)T * 256, 0);
  }

  // ---- residual + outputs ----
  float* outF = (float*)d_out;            // [0:64] x_class, [64:] x2a
  long tx = (long)B * T * 256;
  k_x2a<<<(int)((tx + 255) / 256), 256, 0, stream>>>(attnvF, xc32, f(44), outF + 64, tx);
  k_classifier<<<B, 256, 0, stream>>>(outF + 64, f(45), f(46), outF, T);
}
